// EuclideanCodebook_88510686036439
// MI455X (gfx1250) — matrix operand order repaired and checked by reading, not compiled, not run on hardware
//
#include <hip/hip_runtime.h>

#define CODEBOOK 1024
#define DIM      128
#define NROWS    65536          // 16 * 4096
#define DECAYF   0.8f

typedef __bf16 bf16_t;
typedef bf16_t v16bf __attribute__((ext_vector_type(16)));
typedef float  v8f   __attribute__((ext_vector_type(8)));
typedef float  v4f   __attribute__((ext_vector_type(4)));

// ---------------------------------------------------------------------------
// Kernel 1: prep — bf16 codebook, squared norms, zero accumulators
// grid = CODEBOOK blocks, 128 threads (4 waves)
// ---------------------------------------------------------------------------
__global__ __launch_bounds__(128) void vq_prep(
    const float* __restrict__ embed,
    bf16_t*      __restrict__ embed_bf,
    float*       __restrict__ sqn,
    float*       __restrict__ embed_curr,
    float*       __restrict__ num_curr)
{
    const int n = blockIdx.x;
    const int t = threadIdx.x;
    const float f = embed[(size_t)n * DIM + t];
    embed_bf[(size_t)n * DIM + t]   = (bf16_t)f;
    embed_curr[(size_t)n * DIM + t] = 0.0f;

    float s = f * f;
#pragma unroll
    for (int off = 16; off > 0; off >>= 1)
        s += __shfl_xor(s, off, 32);

    __shared__ float ls[4];
    const int wave = t >> 5, lane = t & 31;
    if (lane == 0) ls[wave] = s;
    __syncthreads();
    if (t == 0) {
        sqn[n]      = ls[0] + ls[1] + ls[2] + ls[3];
        num_curr[n] = 0.0f;
    }
}

// ---------------------------------------------------------------------------
// Helper: convert one 16x32 fp32 sub-tile of x into a 16-bit WMMA A operand, contiguous k order (B matches)
// lane m, half h: VGPR0-7 = K h*16..h*16+15
// ---------------------------------------------------------------------------
__device__ __forceinline__ v16bf load_a_tile(const float* __restrict__ xrow,
                                             int t4, int h)
{
    const int b1 = t4 * 32 + h * 16;
    const int b2 = b1 + 8;
    const v4f f0 = *(const v4f*)(xrow + b1);
    const v4f f1 = *(const v4f*)(xrow + b1 + 4);
    const v4f f2 = *(const v4f*)(xrow + b2);
    const v4f f3 = *(const v4f*)(xrow + b2 + 4);
    v16bf av;
    av[0]=(bf16_t)f0[0];  av[1]=(bf16_t)f0[1];  av[2]=(bf16_t)f0[2];  av[3]=(bf16_t)f0[3];
    av[4]=(bf16_t)f1[0];  av[5]=(bf16_t)f1[1];  av[6]=(bf16_t)f1[2];  av[7]=(bf16_t)f1[3];
    av[8]=(bf16_t)f2[0];  av[9]=(bf16_t)f2[1];  av[10]=(bf16_t)f2[2]; av[11]=(bf16_t)f2[3];
    av[12]=(bf16_t)f3[0]; av[13]=(bf16_t)f3[1]; av[14]=(bf16_t)f3[2]; av[15]=(bf16_t)f3[3];
    return av;
}

// ---------------------------------------------------------------------------
// Kernel 2: WMMA distance + argmin + gather(quantize) + scatter(segment sums)
// 256 threads = 8 waves; each wave owns TWO 16-row tiles (32 rows) so each
// B tile load feeds 8 WMMAs (two independent accumulator chains).
// grid = NROWS / (32*8) = 256 blocks
// ---------------------------------------------------------------------------
__global__ __launch_bounds__(256) void vq_argmin(
    const float*  __restrict__ x,
    const float*  __restrict__ embed,
    const bf16_t* __restrict__ embed_bf,
    const float*  __restrict__ sqn,
    float*        __restrict__ quantize,
    float*        __restrict__ embed_curr,
    float*        __restrict__ num_curr)
{
    const int tid  = threadIdx.x;
    const int lane = tid & 31;
    const int wave = tid >> 5;
    const int h    = lane >> 4;      // K-half selector (A & B 16-bit layouts)
    const int m    = lane & 15;      // row within A tile / col within B tile
    const int row0 = (blockIdx.x * 8 + wave) * 32;

    // ---- A: two 16x128 row tiles of x, fp32 -> bf16 in WMMA A layout ------
    v16bf a0[4], a1[4];
    const float* xrow0 = x + (size_t)(row0 + m) * DIM;
    const float* xrow1 = x + (size_t)(row0 + 16 + m) * DIM;
#pragma unroll
    for (int t4 = 0; t4 < 4; ++t4) {
        a0[t4] = load_a_tile(xrow0, t4, h);
        a1[t4] = load_a_tile(xrow1, t4, h);
    }

    float bestv0[8], bestv1[8];
    int   besti0[8], besti1[8];
#pragma unroll
    for (int j = 0; j < 8; ++j) {
        bestv0[j] = INFINITY; besti0[j] = 0;
        bestv1[j] = INFINITY; besti1[j] = 0;
    }

    // ---- sweep all 64 N-tiles of the codebook -----------------------------
    for (int nt = 0; nt < CODEBOOK / 16; ++nt) {
        const int n = nt * 16 + m;                       // this lane's column
        const bf16_t* brow = embed_bf + (size_t)n * DIM + h * 16;
        if (nt + 1 < CODEBOOK / 16)
            __builtin_prefetch(embed_bf + (size_t)(n + 16) * DIM + h * 16, 0, 3);

        v8f acc0 = {};
        v8f acc1 = {};
#pragma unroll
        for (int t4 = 0; t4 < 4; ++t4) {
            const v16bf bv = *(const v16bf*)(brow + t4 * 32);
            // identical opcode, same B regs next -> cache B operand (reuse_b)
            acc0 = __builtin_amdgcn_wmma_f32_16x16x32_bf16(
                       false, a0[t4], false, bv, (short)0, acc0, false, true);
            acc1 = __builtin_amdgcn_wmma_f32_16x16x32_bf16(
                       false, a1[t4], false, bv, (short)0, acc1, false, false);
        }

        const float c = sqn[n];                          // ||e_n||^2 bias
#pragma unroll
        for (int j = 0; j < 8; ++j) {
            const float d0 = __builtin_fmaf(-2.0f, acc0[j], c);
            const float d1 = __builtin_fmaf(-2.0f, acc1[j], c);
            if (d0 < bestv0[j]) { bestv0[j] = d0; besti0[j] = n; }
            if (d1 < bestv1[j]) { bestv1[j] = d1; besti1[j] = n; }
        }
    }

    // ---- argmin reduce across the 16 lanes sharing each output row --------
#pragma unroll
    for (int j = 0; j < 8; ++j) {
#pragma unroll
        for (int off = 1; off < 16; off <<= 1) {
            float ov; int oi;
            ov = __shfl_xor(bestv0[j], off, 32);
            oi = __shfl_xor(besti0[j], off, 32);
            if (ov < bestv0[j] || (ov == bestv0[j] && oi < besti0[j])) {
                bestv0[j] = ov; besti0[j] = oi;
            }
            ov = __shfl_xor(bestv1[j], off, 32);
            oi = __shfl_xor(besti1[j], off, 32);
            if (ov < bestv1[j] || (ov == bestv1[j] && oi < besti1[j])) {
                bestv1[j] = ov; besti1[j] = oi;
            }
        }
    }
    // tile0: lanes 0-15 hold rows row0+0..7 (j), lanes 16-31 rows row0+8..15
    // tile1: same for rows row0+16..31

    // ---- gather quantize rows; scatter x into segment sums ----------------
    const int d0 = lane * 4;                             // 4 dims per lane
#pragma unroll
    for (int r = 0; r < 32; ++r) {
        const int j   = r & 7;
        const int src = (r >> 3) & 1 ? 16 : 0;           // which lane half
        const int idx = (r < 16) ? __shfl(besti0[j], src, 32)
                                 : __shfl(besti1[j], src, 32);

        const v4f e = *(const v4f*)(embed + (size_t)idx * DIM + d0);
        __builtin_nontemporal_store(e,
            (v4f*)(quantize + (size_t)(row0 + r) * DIM + d0));

        const v4f xv = *(const v4f*)(x + (size_t)(row0 + r) * DIM + d0);
        float* ec = embed_curr + (size_t)idx * DIM + d0;
        atomicAdd(ec + 0, xv[0]);
        atomicAdd(ec + 1, xv[1]);
        atomicAdd(ec + 2, xv[2]);
        atomicAdd(ec + 3, xv[3]);
        if (lane == 0) atomicAdd(num_curr + idx, 1.0f);
    }
}

// ---------------------------------------------------------------------------
// Kernel 3: EMA update + normalize
// grid = CODEBOOK blocks, 128 threads
// ---------------------------------------------------------------------------
__global__ __launch_bounds__(128) void vq_finalize(
    const float* __restrict__ ema_embed,
    const float* __restrict__ ema_num,
    const float* __restrict__ embed_curr,
    const float* __restrict__ num_curr,
    float* __restrict__ out_embed_new,
    float* __restrict__ out_ema_embed,
    float* __restrict__ out_ema_num)
{
    const int n = blockIdx.x;
    const int t = threadIdx.x;
    const float en = ema_num[n] * DECAYF + (1.0f - DECAYF) * num_curr[n];
    const float ee = ema_embed[(size_t)n * DIM + t] * DECAYF
                   + (1.0f - DECAYF) * embed_curr[(size_t)n * DIM + t];
    out_ema_embed[(size_t)n * DIM + t] = ee;
    out_embed_new[(size_t)n * DIM + t] = ee / en;
    if (t == 0) out_ema_num[n] = en;
}

// ---------------------------------------------------------------------------
extern "C" void kernel_launch(void* const* d_in, const int* in_sizes, int n_in,
                              void* d_out, int out_size, void* d_ws, size_t ws_size,
                              hipStream_t stream)
{
    const float* x         = (const float*)d_in[0];
    const float* embed     = (const float*)d_in[1];
    const float* ema_embed = (const float*)d_in[2];
    const float* ema_num   = (const float*)d_in[3];

    float* out            = (float*)d_out;
    float* quantize       = out;                                   // 8,388,608
    float* out_embed_new  = out + (size_t)NROWS * DIM;             //   131,072
    float* out_ema_embed  = out_embed_new + CODEBOOK * DIM;        //   131,072
    float* out_ema_num    = out_ema_embed + CODEBOOK * DIM;        //     1,024

    char* ws = (char*)d_ws;
    bf16_t* embed_bf   = (bf16_t*)(ws + 0);        // 262144 B
    float*  sqn        = (float*) (ws + 262144);   //   4096 B
    float*  embed_curr = (float*) (ws + 266240);   // 524288 B
    float*  num_curr   = (float*) (ws + 790528);   //   4096 B

    vq_prep<<<CODEBOOK, DIM, 0, stream>>>(embed, embed_bf, sqn,
                                          embed_curr, num_curr);
    vq_argmin<<<NROWS / (32 * 8), 256, 0, stream>>>(x, embed, embed_bf, sqn,
                                                    quantize, embed_curr, num_curr);
    vq_finalize<<<CODEBOOK, DIM, 0, stream>>>(ema_embed, ema_num,
                                              embed_curr, num_curr,
                                              out_embed_new, out_ema_embed,
                                              out_ema_num);
}
